// stackedIndRNN_encoder_81956565942493
// MI455X (gfx1250) — matrix operand order repaired and checked by reading, not compiled, not run on hardware
//
#include <hip/hip_runtime.h>
#include <hip/hip_bf16.h>
#include <stdint.h>

typedef __bf16 bf16_t;
typedef __attribute__((ext_vector_type(16))) __bf16 v16bf;
typedef __attribute__((ext_vector_type(8)))  __bf16 v8bf;
typedef __attribute__((ext_vector_type(8)))  float   v8f;

// Fixed problem sizes from the reference
constexpr int T_    = 256;
constexpr int B_    = 128;
constexpr int C_    = 1024;
constexpr int L_    = 6;
constexpr int H_    = 1024;
constexpr int NCLS_ = 1000;
constexpr int Kdim  = 1024;           // reduction dim of every GEMM
constexpr int Mrows = T_ * B_;        // 32768
constexpr float EPS_ = 1e-5f;

__device__ __forceinline__ bf16_t f2bf(float f) {
  union { float f; uint32_t u; } in; in.f = f;
  const uint32_t r = (in.u + 0x7FFFu + ((in.u >> 16) & 1u)) >> 16;  // RNE
  union { uint16_t u; bf16_t b; } out; out.u = (uint16_t)r;
  return out.b;
}

// ---------------------------------------------------------------------------
// WMMA GEMM: C[M, Nout] = A[M,1024] (bf16) x Bt[1024,1024]^T (bf16, stored NxK)
//            + bias, fp32 accumulate/out.
// Block = 256 threads (8 waves), 128x128 tile; wave = 32x64 (2x4 WMMA tiles).
// ---------------------------------------------------------------------------
union FragU { v16bf v; v8bf h[2]; };

__global__ __launch_bounds__(256) void gemm_bf16_wmma(
    const bf16_t* __restrict__ A,    // [M, Kdim] row-major
    const bf16_t* __restrict__ Bt,   // [1024, Kdim] row-major (N x K)
    const float*  __restrict__ bias, // [>= Nout]
    float* __restrict__ C,           // [M, Nout] row-major, ld = Nout
    int Nout)
{
  __shared__ bf16_t sA[128 * 72];    // 72 = 64 + 8 pad -> conflict-free b128 reads
  __shared__ bf16_t sB[128 * 72];

  const int tid   = threadIdx.x;
  const int lane  = tid & 31;
  const int wave  = tid >> 5;
  const int waveM = wave & 3;        // 0..3 (M direction)
  const int waveN = wave >> 2;       // 0..1 (N direction)
  const int lm    = lane & 15;
  const int lh    = lane >> 4;

  const int blockN = blockIdx.x;     // 0..7
  const int blockM = blockIdx.y;

  const int ldRow = tid >> 3;        // 0..31
  const int ldCol = (tid & 7) << 3;  // 0,8,...,56 (bf16 elements)

  const bf16_t* Ag = A  + (size_t)(blockM * 128) * Kdim;
  const bf16_t* Bg = Bt + (size_t)(blockN * 128) * Kdim;

  v8f acc[2][4] = {};

  for (int kb = 0; kb < Kdim; kb += 64) {
    __syncthreads();
    #pragma unroll
    for (int s = 0; s < 4; ++s) {
      const int row = ldRow + s * 32;
      const uint4 va = *(const uint4*)(Ag + (size_t)row * Kdim + kb + ldCol);
      *(uint4*)(&sA[row * 72 + ldCol]) = va;
      const uint4 vb = *(const uint4*)(Bg + (size_t)row * Kdim + kb + ldCol);
      *(uint4*)(&sB[row * 72 + ldCol]) = vb;
    }
    __syncthreads();

    #pragma unroll
    for (int ks = 0; ks < 64; ks += 32) {
      // A fragments: lane holds M = base+lm; K halves 0..7 / 16..23 (lanes 0-15)
      // and 8..15 / 24..31 (lanes 16-31)  [ISA 7.12.2, 16-bit A 16x32]
      FragU af[2];
      #pragma unroll
      for (int mt = 0; mt < 2; ++mt) {
        const int row = waveM * 32 + mt * 16 + lm;
        const bf16_t* p = &sA[row * 72 + ks + lh * 8];
        af[mt].h[0] = *(const v8bf*)(p);
        af[mt].h[1] = *(const v8bf*)(p + 16);
      }
      // B fragments: lane holds N = base+lm; K 0..7 / 16..23 (lanes 0-15), 8..15 / 24..31 (lanes 16-31)
      FragU bfr[4];
      #pragma unroll
      for (int nt = 0; nt < 4; ++nt) {
        const int col = waveN * 64 + nt * 16 + lm;
        const bf16_t* p = &sB[col * 72 + ks + lh * 8];
        bfr[nt].h[0] = *(const v8bf*)(p);
        bfr[nt].h[1] = *(const v8bf*)(p + 16);
      }
      #pragma unroll
      for (int mt = 0; mt < 2; ++mt)
        #pragma unroll
        for (int nt = 0; nt < 4; ++nt)
          acc[mt][nt] = __builtin_amdgcn_wmma_f32_16x16x32_bf16(
              false, af[mt].v, false, bfr[nt].v, (short)0, acc[mt][nt],
              false, false);
    }
  }

  // Epilogue: C/D layout element r <-> (M = r + 8*lh, N = lm)
  #pragma unroll
  for (int nt = 0; nt < 4; ++nt) {
    const int col = blockN * 128 + waveN * 64 + nt * 16 + lm;
    if (col < Nout) {
      const float bv = bias[col];
      #pragma unroll
      for (int mt = 0; mt < 2; ++mt) {
        const int rowbase = blockM * 128 + waveM * 32 + mt * 16 + lh * 8;
        #pragma unroll
        for (int r = 0; r < 8; ++r)
          C[(size_t)(rowbase + r) * Nout + col] = acc[mt][nt][r] + bv;
      }
    }
  }
}

// ---------------------------------------------------------------------------
// Transpose fp32 [R,S] -> bf16 [Spad,R]; zero-fills S..Spad. blockIdx.z batches.
// ---------------------------------------------------------------------------
__global__ void transpose_f32_to_bf16(const float* __restrict__ src,
                                      bf16_t* __restrict__ dst,
                                      int R, int S, int Spad)
{
  src += (size_t)blockIdx.z * R * S;
  dst += (size_t)blockIdx.z * Spad * R;
  __shared__ float tile[32][33];
  const int s0 = blockIdx.x * 32, r0 = blockIdx.y * 32;
  #pragma unroll
  for (int i = 0; i < 4; ++i) {
    const int r = r0 + threadIdx.y + i * 8;
    const int s = s0 + threadIdx.x;
    tile[threadIdx.y + i * 8][threadIdx.x] =
        (r < R && s < S) ? src[(size_t)r * S + s] : 0.0f;
  }
  __syncthreads();
  #pragma unroll
  for (int i = 0; i < 4; ++i) {
    const int s = s0 + threadIdx.y + i * 8;
    const int r = r0 + threadIdx.x;
    if (s < Spad && r < R)
      dst[(size_t)s * R + r] = f2bf(tile[threadIdx.x][threadIdx.y + i * 8]);
  }
}

__global__ void f32_to_bf16_vec(const float* __restrict__ src,
                                bf16_t* __restrict__ dst, int n)
{
  for (int i = blockIdx.x * blockDim.x + threadIdx.x; i < n;
       i += gridDim.x * blockDim.x)
    dst[i] = f2bf(src[i]);
}

// Deterministic two-pass column stats: grid (H/256, 128), 256-row chunks
__global__ void col_partial_stats(const float* __restrict__ Z,
                                  float* __restrict__ psum,
                                  float* __restrict__ psumsq)
{
  const int h  = blockIdx.x * 256 + threadIdx.x;
  const int r0 = blockIdx.y * 256;
  const float* p = Z + (size_t)r0 * H_ + h;
  float s = 0.f, s2 = 0.f;
  for (int r = 0; r < 256; ++r) {
    const float z = p[(size_t)r * H_];
    s += z; s2 += z * z;
  }
  psum  [blockIdx.y * H_ + h] = s;
  psumsq[blockIdx.y * H_ + h] = s2;
}

__global__ void bn_finalize(const float* __restrict__ psum,
                            const float* __restrict__ psumsq,
                            const float* __restrict__ gamma,
                            const float* __restrict__ beta,
                            float* __restrict__ scale,
                            float* __restrict__ shift)
{
  const int h = blockIdx.x * 256 + threadIdx.x;
  float s = 0.f, s2 = 0.f;
  for (int j = 0; j < 128; ++j) { s += psum[j * H_ + h]; s2 += psumsq[j * H_ + h]; }
  const float invN = 1.0f / (float)Mrows;
  const float mean = s * invN;
  const float var  = s2 * invN - mean * mean;
  const float sc   = gamma[h] * rsqrtf(var + EPS_);
  scale[h] = sc;
  shift[h] = beta[h] - mean * sc;
}

// Fused BN-apply + IndRNN recurrence + bf16 convert. One thread per (b,h).
__global__ void indrnn_step(const float* __restrict__ Z,
                            const float* __restrict__ scale,
                            const float* __restrict__ shift,
                            const float* __restrict__ u,
                            bf16_t* __restrict__ Aout)
{
  const int idx = blockIdx.x * 256 + threadIdx.x;    // B_*H_ total
  const int h = idx & (H_ - 1);
  const float sc = scale[h], sh = shift[h], uu = u[h];
  float hp = 0.f;
  size_t off = idx;
  for (int t = 0; t < T_; ++t, off += (size_t)B_ * H_) {
    const float z  = Z[off] * sc + sh;
    const float hn = fmaxf(fmaf(uu, hp, z), 0.f);
    hp = hn;
    Aout[off] = f2bf(hn);
  }
}

// ---------------------------------------------------------------------------
extern "C" void kernel_launch(void* const* d_in, const int* in_sizes, int n_in,
                              void* d_out, int out_size, void* d_ws, size_t ws_size,
                              hipStream_t stream)
{
  (void)in_sizes; (void)n_in; (void)out_size; (void)ws_size;
  const float* x     = (const float*)d_in[0];
  const float* W     = (const float*)d_in[1];
  const float* b     = (const float*)d_in[2];
  const float* gamma = (const float*)d_in[3];
  const float* beta  = (const float*)d_in[4];
  const float* u     = (const float*)d_in[5];
  const float* Wc    = (const float*)d_in[6];
  const float* bc    = (const float*)d_in[7];
  float* out = (float*)d_out;

  char* ws = (char*)d_ws;
  size_t off = 0;
  auto alloc = [&](size_t bytes) -> void* {
    void* p = ws + off;
    off = (off + bytes + 255) & ~(size_t)255;
    return p;
  };
  bf16_t* Abuf   = (bf16_t*)alloc((size_t)Mrows * H_ * sizeof(bf16_t));   // 64 MB
  float*  Zbuf   = (float*) alloc((size_t)Mrows * H_ * sizeof(float));    // 128 MB
  bf16_t* Wt     = (bf16_t*)alloc((size_t)L_ * H_ * C_ * sizeof(bf16_t)); // 12 MB
  bf16_t* Wct    = (bf16_t*)alloc((size_t)1024 * H_ * sizeof(bf16_t));    // 2 MB
  float*  psum   = (float*) alloc((size_t)128 * H_ * sizeof(float));
  float*  psumsq = (float*) alloc((size_t)128 * H_ * sizeof(float));
  float*  scale  = (float*) alloc(H_ * sizeof(float));
  float*  shift  = (float*) alloc(H_ * sizeof(float));

  // Stage weights: W[l] (CxH) -> Wt[l] (HxC) bf16 ; Wc (HxNCLS) -> Wct (1024xH)
  transpose_f32_to_bf16<<<dim3(32, 32, L_), dim3(32, 8), 0, stream>>>(W, Wt, C_, H_, H_);
  transpose_f32_to_bf16<<<dim3(32, 32, 1), dim3(32, 8), 0, stream>>>(Wc, Wct, H_, NCLS_, 1024);
  // x -> bf16 activations
  f32_to_bf16_vec<<<4096, 256, 0, stream>>>(x, Abuf, Mrows * C_);

  for (int l = 0; l < L_; ++l) {
    gemm_bf16_wmma<<<dim3(8, Mrows / 128), 256, 0, stream>>>(
        Abuf, Wt + (size_t)l * H_ * C_, b + l * H_, Zbuf, H_);
    col_partial_stats<<<dim3(H_ / 256, 128), 256, 0, stream>>>(Zbuf, psum, psumsq);
    bn_finalize<<<H_ / 256, 256, 0, stream>>>(psum, psumsq,
                                              gamma + l * H_, beta + l * H_,
                                              scale, shift);
    indrnn_step<<<(B_ * H_) / 256, 256, 0, stream>>>(Zbuf, scale, shift,
                                                     u + l * H_, Abuf);
  }

  // Classifier on last timestep: rows (T-1)*B .. (T-1)*B+127 of Abuf
  const bf16_t* Alast = Abuf + (size_t)(T_ - 1) * B_ * H_;
  gemm_bf16_wmma<<<dim3(8, 1), 256, 0, stream>>>(Alast, Wct, bc, out, NCLS_);
}
